// CrossDomainEquivariantNet_2886218023249
// MI455X (gfx1250) — matrix operand order repaired and checked by reading, not compiled, not run on hardware
//
#include <hip/hip_runtime.h>
#include <hip/hip_bf16.h>
#include <cstdint>

// ---------------------------------------------------------------------------
// CDNA5 (gfx1250) implementation of the PaiNN-like equivariant GNN.
// All 128-wide MLPs run on v_wmma_f32_16x16x32_bf16 (fp32 accum).
// Edge kernel: M=32 edges per wave (2 C-accumulators share each B fragment ->
// 2x arithmetic intensity on the weight stream). Weights pre-transposed to
// [N][K] bf16 so each B fragment is two contiguous 16B runs per lane.
// ---------------------------------------------------------------------------

typedef __attribute__((ext_vector_type(16))) __bf16 v16bf;
typedef __attribute__((ext_vector_type(8)))  float  v8f;

#define HDIM 128
#define VDIM 64
#define CUTOFF 5.0f

__device__ __forceinline__ float silu_f(float x) { return x / (1.0f + __expf(-x)); }
__device__ __forceinline__ float sigm_f(float x) { return 1.0f / (1.0f + __expf(-x)); }

__device__ __forceinline__ v8f wmma_bf16(v16bf a, v16bf b, v8f c) {
  // (neg_a, A, neg_b, B, c_mod, C, reuse_a, reuse_b)
  return __builtin_amdgcn_wmma_f32_16x16x32_bf16(false, a, false, b, (short)0, c,
                                                 false, false);
}

// A fragment (16x32, MxK) from a row-major LDS tile with leading dim ld.
// Layout per ISA: lanes 0-15 -> M=lane, K offsets {kb..kb+7, kb+16..kb+23},
// kb = 0 for lanes 0-15, 8 for lanes 16-31.
__device__ __forceinline__ v16bf a_frag_lds(const __bf16* tile, int ld, int k0) {
  int lane = threadIdx.x & 31;
  int m  = lane & 15;
  int kb = k0 + ((lane & 16) ? 8 : 0);
  const __bf16* p = tile + m * ld + kb;
  v16bf a;
#pragma unroll
  for (int i = 0; i < 8; ++i) { a[i] = p[i]; a[8 + i] = p[16 + i]; }
  return a;
}

// A fragment gathered from the global bf16 node table using 16 per-row node ids.
__device__ __forceinline__ v16bf a_frag_gather(const __bf16* hbt, const int* nodes,
                                               int k0) {
  int lane = threadIdx.x & 31;
  int m  = lane & 15;
  int kb = k0 + ((lane & 16) ? 8 : 0);
  const __bf16* p = hbt + (size_t)nodes[m] * HDIM + kb;
  v16bf a;
#pragma unroll
  for (int i = 0; i < 8; ++i) { a[i] = p[i]; a[8 + i] = p[16 + i]; }
  return a;
}

// B fragment (32x16, KxN) from transposed weights Wt[N][K] (bf16, leading dim K).
// Lane n = lane&15 holds column n; K offsets {kb..kb+7, kb+16..kb+23},
// kb = k0 for lanes 0-15, k0+8 for lanes 16-31 (matches A fragment).
__device__ __forceinline__ v16bf b_frag_t(const __bf16* Wt, int K, int k0, int n0) {
  int lane = threadIdx.x & 31;
  int n  = n0 + (lane & 15);
  int kb = k0 + ((lane & 16) ? 8 : 0);
  const __bf16* p = Wt + (size_t)n * K + kb;
  v16bf b;
#pragma unroll
  for (int i = 0; i < 8; ++i) { b[i] = p[i]; b[8 + i] = p[16 + i]; }
  return b;
}

// hidden = silu(x @ W1t + b1), x: LDS [16][ldx] bf16, out: LDS [16][128] bf16
__device__ __forceinline__ void mlp_hidden(const __bf16* x, int ldx, int K,
                                           const __bf16* W1t, const float* b1,
                                           __bf16* hid) {
  int lane = threadIdx.x & 31;
  int mlo  = (lane & 16) ? 8 : 0;
  int ncol = lane & 15;
  int K32  = K >> 5;
#pragma unroll 1
  for (int nt = 0; nt < 8; ++nt) {
    v8f c = {};
    for (int kt = 0; kt < K32; ++kt) {
      v16bf a = a_frag_lds(x, ldx, kt * 32);
      v16bf b = b_frag_t(W1t, K, kt * 32, nt * 16);
      c = wmma_bf16(a, b, c);
    }
    int n = nt * 16 + ncol;
    float bb = b1[n];
#pragma unroll
    for (int r = 0; r < 8; ++r)
      hid[(mlo + r) * HDIM + n] = (__bf16)silu_f(c[r] + bb);
  }
}

// ---------------------------------------------------------------------------
// weight conversion: W [K][Nout] fp32 row-major -> Wt [Nout][Kpad] bf16
// ---------------------------------------------------------------------------
__global__ void w_to_bf16_t(const float* __restrict__ W, __bf16* __restrict__ Wt,
                            int K, int Kpad, int Nout) {
  int t = blockIdx.x * blockDim.x + threadIdx.x;
  int total = Nout * Kpad;
  if (t >= total) return;
  int n = t / Kpad, k = t % Kpad;
  float v = (k < K) ? W[(size_t)k * Nout + n] : 0.0f;
  Wt[t] = (__bf16)v;
}

// ---------------------------------------------------------------------------
// init: h = embed[z], bf16 mirror
// ---------------------------------------------------------------------------
__global__ void init_nodes(const int* __restrict__ z, const float* __restrict__ embed,
                           float* __restrict__ h, __bf16* __restrict__ hb, int N) {
  int t = blockIdx.x * blockDim.x + threadIdx.x;
  if (t >= N * HDIM) return;
  int node = t >> 7, c = t & 127;
  float v = embed[(size_t)z[node] * HDIM + c];
  h[t] = v;
  hb[t] = (__bf16)v;
}

// ---------------------------------------------------------------------------
// Edge kernel: 4 waves / block, 32 edges per wave (M-blocked x2).
//   msg = MLP_edge([h_dst, h_src, d]) * cutoff(d); scatter-add to h_msg, V_msg
// ---------------------------------------------------------------------------
__global__ __launch_bounds__(128) void edge_kernel(
    const __bf16* __restrict__ hb, const float* __restrict__ pos,
    const int* __restrict__ ei,                 // [2][E] (src row, dst row)
    const __bf16* __restrict__ W1t,             // [128][288] bf16 (K=257 padded)
    const float* __restrict__ b1,
    const __bf16* __restrict__ W2t,             // [128][128] bf16
    const float* __restrict__ b2,
    float* __restrict__ h_msg, float* __restrict__ V_msg, int E) {
  __shared__ int   s_ni[4][32], s_nj[4][32];
  __shared__ float s_cut[4][32], s_d[4][32], s_dir[4][32][3];
  __shared__ __bf16 s_hid[4][32][HDIM];

  int wave = threadIdx.x >> 5;
  int lane = threadIdx.x & 31;
  int e0   = (blockIdx.x * 4 + wave) * 32;

  // one edge per lane: full-wave geometry prologue
  {
    int e = e0 + lane;
    bool valid = e < E;
    int ec = valid ? e : (E - 1);
    int sj = ei[ec];       // source j
    int di = ei[E + ec];   // target i
    float rx = pos[sj * 3 + 0] - pos[di * 3 + 0];
    float ry = pos[sj * 3 + 1] - pos[di * 3 + 1];
    float rz = pos[sj * 3 + 2] - pos[di * 3 + 2];
    float d  = sqrtf(rx * rx + ry * ry + rz * rz);
    float x  = fminf(fmaxf(d * (1.0f / CUTOFF), 0.0f), 1.0f);
    float x2 = x * x, x3 = x2 * x;
    float cut = 1.0f - 6.0f * x3 * x2 + 15.0f * x2 * x2 - 10.0f * x3;
    if (!valid) cut = 0.0f;
    float inv = 1.0f / (d + 1e-8f);
    s_ni[wave][lane] = di;  s_nj[wave][lane] = sj;
    s_cut[wave][lane] = cut; s_d[wave][lane] = d;
    s_dir[wave][lane][0] = rx * inv;
    s_dir[wave][lane][1] = ry * inv;
    s_dir[wave][lane][2] = rz * inv;
  }
  __builtin_amdgcn_wave_barrier();   // scheduling fence; LDS is in-order per wave

  int mlo  = (lane & 16) ? 8 : 0;
  int ncol = lane & 15;

  // hint the second-layer weights into cache while we grind layer 1
  __builtin_prefetch(W2t + (size_t)ncol * HDIM, 0, 1);

  // ---- layer 1: [h_dst | h_src | d] (K=257) @ W1, + b1, SiLU -> LDS ----
#pragma unroll 1
  for (int nt = 0; nt < 8; ++nt) {
    v8f c0 = {}, c1 = {};
#pragma unroll
    for (int kt = 0; kt < 4; ++kt) {               // h_dst part, K 0..127
      v16bf b = b_frag_t(W1t, 288, kt * 32, nt * 16);
      c0 = wmma_bf16(a_frag_gather(hb, &s_ni[wave][0],  kt * 32), b, c0);
      c1 = wmma_bf16(a_frag_gather(hb, &s_ni[wave][16], kt * 32), b, c1);
    }
#pragma unroll
    for (int kt = 0; kt < 4; ++kt) {               // h_src part, K 128..255
      v16bf b = b_frag_t(W1t, 288, 128 + kt * 32, nt * 16);
      c0 = wmma_bf16(a_frag_gather(hb, &s_nj[wave][0],  kt * 32), b, c0);
      c1 = wmma_bf16(a_frag_gather(hb, &s_nj[wave][16], kt * 32), b, c1);
    }
    int n = nt * 16 + ncol;
    float w1d = (float)W1t[(size_t)n * 288 + 256];  // distance column (rank-1)
    float bb  = b1[n];
#pragma unroll
    for (int r = 0; r < 8; ++r) {
      int m = mlo + r;
      s_hid[wave][m][n]      = (__bf16)silu_f(c0[r] + bb + s_d[wave][m] * w1d);
      s_hid[wave][16 + m][n] = (__bf16)silu_f(c1[r] + bb + s_d[wave][16 + m] * w1d);
    }
  }
  __builtin_amdgcn_wave_barrier();

  // ---- layer 2: hidden @ W2 + b2, * cutoff, scatter-add ----
#pragma unroll 1
  for (int nt = 0; nt < 8; ++nt) {
    v8f c0 = {}, c1 = {};
#pragma unroll
    for (int kt = 0; kt < 4; ++kt) {
      v16bf b = b_frag_t(W2t, 128, kt * 32, nt * 16);
      c0 = wmma_bf16(a_frag_lds(&s_hid[wave][0][0],  HDIM, kt * 32), b, c0);
      c1 = wmma_bf16(a_frag_lds(&s_hid[wave][16][0], HDIM, kt * 32), b, c1);
    }
    int n = nt * 16 + ncol;
    float bb = b2[n];
#pragma unroll
    for (int half = 0; half < 2; ++half) {
      v8f c = half ? c1 : c0;
#pragma unroll
      for (int r = 0; r < 8; ++r) {
        int m = half * 16 + mlo + r;
        float val = (c[r] + bb) * s_cut[wave][m];  // tail edges: cut==0 -> add 0
        int node = s_ni[wave][m];
        atomicAdd(&h_msg[(size_t)node * HDIM + n], val);
        if (n < VDIM) {
          size_t vb = (size_t)node * 3 * VDIM + n;
          atomicAdd(&V_msg[vb + 0 * VDIM], s_dir[wave][m][0] * val);
          atomicAdd(&V_msg[vb + 1 * VDIM], s_dir[wave][m][1] * val);
          atomicAdd(&V_msg[vb + 2 * VDIM], s_dir[wave][m][2] * val);
        }
      }
    }
  }
}

// ---------------------------------------------------------------------------
// Node kernel: 4 waves / block, 16 nodes per wave. Runs the full per-node
// update chain (node MLP, vgate MLP, vector norm, norm MLP, gate MLP).
// ---------------------------------------------------------------------------
__global__ __launch_bounds__(128) void node_kernel(
    float* __restrict__ h, __bf16* __restrict__ hb, float* __restrict__ V,
    const float* __restrict__ h_msg, const float* __restrict__ V_msg,
    const __bf16* n1t, const float* n_b1, const __bf16* n2t, const float* n_b2,
    const __bf16* v1t, const float* v_b1, const __bf16* v2t, const float* v_b2,
    const __bf16* no1t, const float* no_b1, const __bf16* no2t, const float* no_b2,
    const __bf16* g1t, const float* g_b1, const __bf16* g2t, const float* g_b2,
    int N) {
  __shared__ __bf16 s_x[4][16][192];
  __shared__ __bf16 s_hid[4][16][HDIM];
  __shared__ int    s_node[4][16];

  int wave = threadIdx.x >> 5;
  int lane = threadIdx.x & 31;
  __bf16 (*xw)[192] = s_x[wave];
  __bf16* hidw = &s_hid[wave][0][0];
  int base = (blockIdx.x * 4 + wave) * 16;

  if (lane < 16) {
    int nd = base + lane;
    s_node[wave][lane] = (nd < N) ? nd : (N - 1);
  }
  __builtin_amdgcn_wave_barrier();

  int mlo  = (lane & 16) ? 8 : 0;
  int ncol = lane & 15;

  // stage x[:, :128] <- bf16(h_msg rows)
  {
    int m = lane & 15, c0 = (lane & 16) ? 64 : 0;
    const float* src = h_msg + (size_t)s_node[wave][m] * HDIM + c0;
    for (int t = 0; t < 64; ++t) xw[m][c0 + t] = (__bf16)src[t];
  }
  __builtin_amdgcn_wave_barrier();

  // ---- h = h + MLP_node(h_msg) ----
  mlp_hidden(&xw[0][0], 192, 128, n1t, n_b1, hidw);
#pragma unroll 1
  for (int nt = 0; nt < 8; ++nt) {
    v8f c = {};
#pragma unroll
    for (int kt = 0; kt < 4; ++kt)
      c = wmma_bf16(a_frag_lds(hidw, HDIM, kt * 32),
                    b_frag_t(n2t, 128, kt * 32, nt * 16), c);
    int n = nt * 16 + ncol;
    float bb = n_b2[n];
#pragma unroll
    for (int r = 0; r < 8; ++r) {
      int m = mlo + r;
      if (base + m < N) {
        size_t off = (size_t)s_node[wave][m] * HDIM + n;
        float hv = h[off] + c[r] + bb;
        h[off] = hv;
        hb[off] = (__bf16)hv;
      }
    }
  }

  // ---- V = V + V_msg * sigmoid(MLP_vgate(h_msg)); Vn -> x[:,128:192] ----
  mlp_hidden(&xw[0][0], 192, 128, v1t, v_b1, hidw);
#pragma unroll 1
  for (int nt = 0; nt < 4; ++nt) {
    v8f c = {};
#pragma unroll
    for (int kt = 0; kt < 4; ++kt)
      c = wmma_bf16(a_frag_lds(hidw, HDIM, kt * 32),
                    b_frag_t(v2t, 128, kt * 32, nt * 16), c);
    int n = nt * 16 + ncol;
    float bb = v_b2[n];
#pragma unroll
    for (int r = 0; r < 8; ++r) {
      int m = mlo + r;
      int nd = s_node[wave][m];
      bool valid = (base + m) < N;
      float g = sigm_f(c[r] + bb);
      float vn2 = 1e-12f;
#pragma unroll
      for (int c3 = 0; c3 < 3; ++c3) {
        size_t off = (size_t)nd * 3 * VDIM + c3 * VDIM + n;
        float vv = V[off] + V_msg[off] * g;
        if (valid) V[off] = vv;
        vn2 += vv * vv;
      }
      xw[m][HDIM + n] = (__bf16)sqrtf(vn2);
    }
  }
  __builtin_amdgcn_wave_barrier();

  // reload x[:, :128] <- bf16(h_new)
  {
    int m = lane & 15, c0 = (lane & 16) ? 64 : 0;
    const __bf16* src = hb + (size_t)s_node[wave][m] * HDIM + c0;
    for (int t = 0; t < 64; ++t) xw[m][c0 + t] = src[t];
  }
  __builtin_amdgcn_wave_barrier();

  // ---- h = h + MLP_norm([h, Vn]) (K=192) ----
  mlp_hidden(&xw[0][0], 192, 192, no1t, no_b1, hidw);
#pragma unroll 1
  for (int nt = 0; nt < 8; ++nt) {
    v8f c = {};
#pragma unroll
    for (int kt = 0; kt < 4; ++kt)
      c = wmma_bf16(a_frag_lds(hidw, HDIM, kt * 32),
                    b_frag_t(no2t, 128, kt * 32, nt * 16), c);
    int n = nt * 16 + ncol;
    float bb = no_b2[n];
#pragma unroll
    for (int r = 0; r < 8; ++r) {
      int m = mlo + r;
      size_t off = (size_t)s_node[wave][m] * HDIM + n;
      float hv = h[off] + c[r] + bb;
      if (base + m < N) { h[off] = hv; hb[off] = (__bf16)hv; }
      xw[m][n] = (__bf16)hv;   // gate-MLP input
    }
  }
  __builtin_amdgcn_wave_barrier();

  // ---- V *= sigmoid(MLP_gate(h)) ----
  mlp_hidden(&xw[0][0], 192, 128, g1t, g_b1, hidw);
#pragma unroll 1
  for (int nt = 0; nt < 4; ++nt) {
    v8f c = {};
#pragma unroll
    for (int kt = 0; kt < 4; ++kt)
      c = wmma_bf16(a_frag_lds(hidw, HDIM, kt * 32),
                    b_frag_t(g2t, 128, kt * 32, nt * 16), c);
    int n = nt * 16 + ncol;
    float bb = g_b2[n];
#pragma unroll
    for (int r = 0; r < 8; ++r) {
      int m = mlo + r;
      if (base + m < N) {
        int nd = s_node[wave][m];
        float g = sigm_f(c[r] + bb);
#pragma unroll
        for (int c3 = 0; c3 < 3; ++c3) {
          size_t off = (size_t)nd * 3 * VDIM + c3 * VDIM + n;
          V[off] *= g;
        }
      }
    }
  }
}

// ---------------------------------------------------------------------------
// readout: atom_pred = MLP_out(h) (128->128->1); molecule segment mean
// ---------------------------------------------------------------------------
__global__ void out_kernel(const float* __restrict__ h, const float* __restrict__ W1,
                           const float* __restrict__ b1, const float* __restrict__ W2,
                           const float* __restrict__ b2, const int* __restrict__ batch,
                           float* __restrict__ atom_pred, float* __restrict__ molsum,
                           float* __restrict__ molcnt, int N) {
  int i = blockIdx.x * blockDim.x + threadIdx.x;
  if (i >= N) return;
  const float* hr = h + (size_t)i * HDIM;
  float out = b2[0];
  for (int j = 0; j < HDIM; ++j) {
    float acc = b1[j];
    for (int k = 0; k < HDIM; ++k) acc += hr[k] * W1[(size_t)k * HDIM + j];
    out += silu_f(acc) * W2[j];
  }
  atom_pred[i] = out;
  atomicAdd(&molsum[batch[i]], out);
  atomicAdd(&molcnt[batch[i]], 1.0f);
}

__global__ void finalize_mol(const float* __restrict__ molsum,
                             const float* __restrict__ molcnt,
                             float* __restrict__ out, int G) {
  int g = blockIdx.x * blockDim.x + threadIdx.x;
  if (g < G) out[g] = molsum[g] / fmaxf(molcnt[g], 1.0f);
}

// ---------------------------------------------------------------------------
// host launcher
// ---------------------------------------------------------------------------
extern "C" void kernel_launch(void* const* d_in, const int* in_sizes, int n_in,
                              void* d_out, int out_size, void* d_ws, size_t ws_size,
                              hipStream_t stream) {
  (void)n_in; (void)ws_size;
  const int LAY = 5;
  int ii = 0;
  const int*   z     = (const int*)d_in[ii++];
  const float* pos   = (const float*)d_in[ii++];
  const int*   ei    = (const int*)d_in[ii++];
  const int*   batch = (const int*)d_in[ii++];
  int N = in_sizes[0];
  int E = in_sizes[2] / 2;
  int G = out_size - N;
  const float* embed = (const float*)d_in[ii++];

  struct Mlp { const float *W1, *b1, *W2, *b2; };
  auto grab = [&]() { Mlp m; m.W1 = (const float*)d_in[ii++]; m.b1 = (const float*)d_in[ii++];
                      m.W2 = (const float*)d_in[ii++]; m.b2 = (const float*)d_in[ii++]; return m; };
  // setup_inputs() insertion order: per layer {edge, node, vgate, norm, gate},
  // each MLP {W1, b1, W2, b2}; then "out".
  Mlp edgeM[LAY], nodeM[LAY], vgateM[LAY], normM[LAY], gateM[LAY];
  for (int l = 0; l < LAY; ++l) {
    edgeM[l] = grab(); nodeM[l] = grab(); vgateM[l] = grab();
    normM[l] = grab(); gateM[l] = grab();
  }
  Mlp outM = grab();

  // workspace carve (256B aligned)
  uintptr_t p = (uintptr_t)d_ws;
  auto carve = [&](size_t bytes) { void* r = (void*)p; p += (bytes + 255) & ~(size_t)255; return r; };
  float*  h      = (float*)carve((size_t)N * HDIM * 4);
  __bf16* hb     = (__bf16*)carve((size_t)N * HDIM * 2);
  float*  V      = (float*)carve((size_t)N * 3 * VDIM * 4);
  float*  h_msg  = (float*)carve((size_t)N * HDIM * 4);
  float*  V_msg  = (float*)carve((size_t)N * 3 * VDIM * 4);
  float*  molsum = (float*)carve((size_t)G * 4);
  float*  molcnt = (float*)carve((size_t)G * 4);
  static const size_t wsz[10] = {128 * 288, 128 * 128, 128 * 128, 128 * 128, 128 * 128,
                                 64 * 128, 128 * 192, 128 * 128, 128 * 128, 64 * 128};
  __bf16* wt[LAY][10];
  for (int l = 0; l < LAY; ++l)
    for (int j = 0; j < 10; ++j) wt[l][j] = (__bf16*)carve(wsz[j] * 2);

  auto tconv = [&](const float* W, __bf16* Wt, int K, int Kpad, int Nout) {
    int total = Nout * Kpad;
    w_to_bf16_t<<<(total + 255) / 256, 256, 0, stream>>>(W, Wt, K, Kpad, Nout);
  };
  for (int l = 0; l < LAY; ++l) {
    tconv(edgeM[l].W1,  wt[l][0], 257, 288, 128);
    tconv(edgeM[l].W2,  wt[l][1], 128, 128, 128);
    tconv(nodeM[l].W1,  wt[l][2], 128, 128, 128);
    tconv(nodeM[l].W2,  wt[l][3], 128, 128, 128);
    tconv(vgateM[l].W1, wt[l][4], 128, 128, 128);
    tconv(vgateM[l].W2, wt[l][5], 128, 128, 64);
    tconv(normM[l].W1,  wt[l][6], 192, 192, 128);
    tconv(normM[l].W2,  wt[l][7], 128, 128, 128);
    tconv(gateM[l].W1,  wt[l][8], 128, 128, 128);
    tconv(gateM[l].W2,  wt[l][9], 128, 128, 64);
  }

  hipMemsetAsync(V, 0, (size_t)N * 3 * VDIM * 4, stream);
  init_nodes<<<(N * HDIM + 255) / 256, 256, 0, stream>>>(z, embed, h, hb, N);

  for (int l = 0; l < LAY; ++l) {
    hipMemsetAsync(h_msg, 0, (size_t)N * HDIM * 4, stream);
    hipMemsetAsync(V_msg, 0, (size_t)N * 3 * VDIM * 4, stream);
    edge_kernel<<<(E + 127) / 128, 128, 0, stream>>>(
        hb, pos, ei, wt[l][0], edgeM[l].b1, wt[l][1], edgeM[l].b2, h_msg, V_msg, E);
    node_kernel<<<(N + 63) / 64, 128, 0, stream>>>(
        h, hb, V, h_msg, V_msg,
        wt[l][2], nodeM[l].b1,  wt[l][3], nodeM[l].b2,
        wt[l][4], vgateM[l].b1, wt[l][5], vgateM[l].b2,
        wt[l][6], normM[l].b1,  wt[l][7], normM[l].b2,
        wt[l][8], gateM[l].b1,  wt[l][9], gateM[l].b2, N);
  }

  float* outF = (float*)d_out;
  hipMemsetAsync(molsum, 0, (size_t)G * 4, stream);
  hipMemsetAsync(molcnt, 0, (size_t)G * 4, stream);
  out_kernel<<<(N + 127) / 128, 128, 0, stream>>>(h, outM.W1, outM.b1, outM.W2, outM.b2,
                                                  batch, outF + G, molsum, molcnt, N);
  finalize_mol<<<(G + 127) / 128, 128, 0, stream>>>(molsum, molcnt, outF, G);
}
